// SoftBiasTransformer_74620761801004
// MI455X (gfx1250) — matrix operand order repaired and checked by reading, not compiled, not run on hardware
//
#include <hip/hip_runtime.h>
#include <hip/hip_bf16.h>
#include <math.h>
#include <stdint.h>

typedef __bf16 bf16;
typedef __attribute__((ext_vector_type(16))) __bf16 v16bf;
typedef __attribute__((ext_vector_type(8)))  float  v8f;
typedef unsigned int u32;
typedef __attribute__((ext_vector_type(4))) unsigned int u32x4;
typedef __attribute__((ext_vector_type(8))) int i32x8;
typedef __attribute__((ext_vector_type(4))) int i32x4;

#define DEV __device__ __forceinline__

#if __has_builtin(__builtin_amdgcn_tensor_load_to_lds)
#define HAS_TDM 1
#else
#define HAS_TDM 0
#endif

constexpr int Bb = 1024, Ss = 64, Dd = 768, Hh = 6, HDh = 128, FFf = 3072;
constexpr int Mrows = Bb * Ss;   // 65536 token rows

// ---------------------------------------------------------------------------
// WMMA fragment helpers (CDNA5 wave32, 16x16x32 bf16, f32 accum)
// A (16x32): lane = {m = lane&15, half = lane>>4}; VGPR v packs K pair
//   k = (v>>2)*16 + half*8 + (v&3)*2 -> per lane: 16B at +0 and 16B at +16 elems
// B (32x16) stored K-contiguous per column n ("BT" layout [n][k]):
//   lane = {n = lane&15, half}; same K order as A -> 16B at +half*8 and 16B at +16
// C/D (16x16): lane&15 = n; VGPR r -> m = r + 8*(lane>>4)
// ---------------------------------------------------------------------------
DEV v16bf load_fragA(const bf16* tile, int ld, int lane) {
  const int m = lane & 15, h = (lane >> 4) & 1;
  const bf16* p = tile + m * ld + h * 8;
  union { v16bf v; int4 q[2]; } u;
  u.q[0] = *reinterpret_cast<const int4*>(p);
  u.q[1] = *reinterpret_cast<const int4*>(p + 16);
  return u.v;
}

DEV v16bf load_fragB(const bf16* tileT, int ld, int lane) {
  const int n = lane & 15, h = (lane >> 4) & 1;
  const bf16* p = tileT + n * ld + h * 8;
  union { v16bf v; int4 q[2]; } u;
  u.q[0] = *reinterpret_cast<const int4*>(p);
  u.q[1] = *reinterpret_cast<const int4*>(p + 16);
  return u.v;
}

DEV v8f wmma_bf16(v16bf a, v16bf b, v8f c) {
  return __builtin_amdgcn_wmma_f32_16x16x32_bf16(false, a, false, b, (short)0, c,
                                                 false, false);
}

#if HAS_TDM
// ---------------------------------------------------------------------------
// TDM: DMA one 128-row x 32-elem bf16 tile (row stride K elems) into LDS,
// padding each 64B row by 16B so LDS rows are 40 bf16 (16B-aligned frags).
// D# per CDNA5 ISA ch.8: group0 = {count/flags, lds_addr, gaddr_lo, gaddr_hi|type},
// group1 = {dsize/pad, dims, tile dims, strides}; groups 2/3 unused (2D tensor).
// 6-arg builtin form: (g0, g1, g2, g3, g4, cpol).
// ---------------------------------------------------------------------------
DEV void tdm_load_tile(const bf16* gsrc, u32 lds_addr, int K) {
  const unsigned long long ga = (unsigned long long)(uintptr_t)gsrc;
  u32x4 g0;
  g0[0] = 1u;                                              // count=1, user desc
  g0[1] = lds_addr;                                        // LDS byte address
  g0[2] = (u32)ga;                                         // global addr [31:0]
  g0[3] = (u32)((ga >> 32) & 0x01FFFFFFu) | (2u << 30);    // addr [56:32] | type=2
  i32x8 g1;
  g1[0] = (int)((1u << 16) |          // data_size = 2B
                (1u << 20) |          // pad_enable
                (3u << 22) |          // pad_interval: 16 DWORDs (64B)
                (3u << 25));          // pad_amount:   4 DWORDs (16B)
  g1[1] = (int)((u32)K << 16);        // tensor_dim0 low16 (K fits 16 bits)
  g1[2] = (int)(128u << 16);          // tensor_dim0 hi=0, tensor_dim1 = 128
  g1[3] = (int)(32u << 16);           // tensor_dim1 hi=0, tile_dim0 = 32
  g1[4] = 128;                        // tile_dim1 = 128, tile_dim2 = 0
  g1[5] = K;                          // tensor_dim0_stride low32
  g1[6] = 0;                          // stride hi / dim1_stride
  g1[7] = 0;
  i32x4 gz4; gz4[0] = gz4[1] = gz4[2] = gz4[3] = 0;
  i32x8 gz8;
  gz8[0] = gz8[1] = gz8[2] = gz8[3] = 0;
  gz8[4] = gz8[5] = gz8[6] = gz8[7] = 0;
  __builtin_amdgcn_tensor_load_to_lds(g0, g1, gz4, gz4, gz8, 0);
}
#endif

// ---------------------------------------------------------------------------
// GEMM: Out[M,N] = A[M,K](bf16) @ W[K,N] + bias, W pre-transposed bf16 Wt[N][K].
// EPI: 0 -> bf16 out, 1 -> bf16 relu out, 2 -> f32 out.
// 128x128x32 tiles, 256 threads (8 waves), wave = 2x4 C-tiles.
// Tiles double-buffered in LDS; staged by TDM (wave 0) overlapping WMMA.
// ---------------------------------------------------------------------------
template <int EPI>
__global__ __launch_bounds__(256) void gemm_kernel(
    const bf16* __restrict__ Ab, const bf16* __restrict__ Wt,
    const float* __restrict__ bias, void* __restrict__ Outv,
    int M, int N, int K) {
  __shared__ __align__(16) bf16 As[2][128][40];
  __shared__ __align__(16) bf16 Bs[2][128][40];

  const int tid  = threadIdx.x;
  const int lane = tid & 31;
  const int w    = tid >> 5;
  const int wm   = w & 3;        // 32-row band
  const int wn   = w >> 2;       // 64-col band
  const int m0   = blockIdx.y * 128;
  const int n0   = blockIdx.x * 128;
  const int KT   = K >> 5;
  (void)M;

  v8f acc[2][4] = {};

#if HAS_TDM
  if (w == 0) {
    tdm_load_tile(Ab + (size_t)m0 * K, (u32)(uintptr_t)&As[0][0][0], K);
    tdm_load_tile(Wt + (size_t)n0 * K, (u32)(uintptr_t)&Bs[0][0][0], K);
  }
#else
  {
    const int kb = 0;
#pragma unroll
    for (int i = 0; i < 2; ++i) {
      const int g = i * 256 + tid, r = g >> 2, c8 = (g & 3) * 8;
      *reinterpret_cast<int4*>(&As[0][r][c8]) =
          *reinterpret_cast<const int4*>(Ab + (size_t)(m0 + r) * K + kb + c8);
      *reinterpret_cast<int4*>(&Bs[0][r][c8]) =
          *reinterpret_cast<const int4*>(Wt + (size_t)(n0 + r) * K + kb + c8);
    }
  }
#endif

  for (int kt = 0; kt < KT; ++kt) {
    const int cur = kt & 1;
#if HAS_TDM
    if (w == 0) {
      if (kt + 1 < KT) {
        const int kb = (kt + 1) << 5;
        tdm_load_tile(Ab + (size_t)m0 * K + kb, (u32)(uintptr_t)&As[cur ^ 1][0][0], K);
        tdm_load_tile(Wt + (size_t)n0 * K + kb, (u32)(uintptr_t)&Bs[cur ^ 1][0][0], K);
        __builtin_amdgcn_s_wait_tensorcnt(2);  // tile kt landed; kt+1 in flight
      } else {
        __builtin_amdgcn_s_wait_tensorcnt(0);
      }
    }
#else
    if (kt + 1 < KT) {
      const int kb = (kt + 1) << 5;
#pragma unroll
      for (int i = 0; i < 2; ++i) {
        const int g = i * 256 + tid, r = g >> 2, c8 = (g & 3) * 8;
        *reinterpret_cast<int4*>(&As[cur ^ 1][r][c8]) =
            *reinterpret_cast<const int4*>(Ab + (size_t)(m0 + r) * K + kb + c8);
        *reinterpret_cast<int4*>(&Bs[cur ^ 1][r][c8]) =
            *reinterpret_cast<const int4*>(Wt + (size_t)(n0 + r) * K + kb + c8);
      }
    }
#endif
    __syncthreads();  // tile `cur` visible to all waves

    v16bf a[2], b[4];
#pragma unroll
    for (int t = 0; t < 2; ++t)
      a[t] = load_fragA(&As[cur][wm * 32 + t * 16][0], 40, lane);
#pragma unroll
    for (int u = 0; u < 4; ++u)
      b[u] = load_fragB(&Bs[cur][wn * 64 + u * 16][0], 40, lane);
#pragma unroll
    for (int t = 0; t < 2; ++t)
#pragma unroll
      for (int u = 0; u < 4; ++u)
        acc[t][u] = wmma_bf16(a[t], b[u], acc[t][u]);
    __syncthreads();  // tile `cur` free for reuse
  }

  const int half = lane >> 4, nl = lane & 15;
#pragma unroll
  for (int t = 0; t < 2; ++t) {
#pragma unroll
    for (int u = 0; u < 4; ++u) {
      const int col = n0 + wn * 64 + u * 16 + nl;
      const float bv = bias[col];
#pragma unroll
      for (int r = 0; r < 8; ++r) {
        const int row = m0 + wm * 32 + t * 16 + half * 8 + r;
        float v = acc[t][u][r] + bv;
        if (EPI == 0) {
          ((bf16*)Outv)[(size_t)row * N + col] = (bf16)v;
        } else if (EPI == 1) {
          ((bf16*)Outv)[(size_t)row * N + col] = (bf16)fmaxf(v, 0.f);
        } else {
          ((float*)Outv)[(size_t)row * N + col] = v;
        }
      }
    }
  }
}

// ---------------------------------------------------------------------------
// Attention, one block per (batch, head). S=64, HD=128.
// scores = QK^T*scale + bias; softmax; ctx = attn @ V. Ctx may alias Q.
// WMMA A/B fragments loaded directly from global (all 16B-aligned).
// ---------------------------------------------------------------------------
__global__ __launch_bounds__(256) void attn_kernel(
    const bf16* __restrict__ Q, const bf16* __restrict__ Km,
    const bf16* __restrict__ V, const float* __restrict__ bias64,
    bf16* __restrict__ Ctx) {
  __shared__ float sc[64][65];
  __shared__ __align__(16) bf16 attS[64][72];
  __shared__ __align__(16) bf16 vt[128][72];

  const int tid  = threadIdx.x;
  const int lane = tid & 31;
  const int w    = tid >> 5;
  const int b    = blockIdx.x / Hh;
  const int h    = blockIdx.x % Hh;
  const size_t base = ((size_t)b * Ss) * Dd + (size_t)h * HDh;

  // transpose V slice into LDS: vt[d][j] = V[j][d]
  {
    const int d = tid & 127, jg = tid >> 7;
    for (int j = jg * 32; j < jg * 32 + 32; ++j)
      vt[d][j] = V[base + (size_t)j * Dd + d];
  }

  // scores: 4x4 tile grid; wave w owns (tm = w&3, tn = 2*(w>>2)+{0,1})
  {
    const int tm = w & 3, tn0 = (w >> 2) * 2;
    v8f acc[2] = {};
#pragma unroll
    for (int k0 = 0; k0 < HDh; k0 += 32) {
      v16bf a = load_fragA(Q + base + (size_t)(tm * 16) * Dd + k0, Dd, lane);
#pragma unroll
      for (int u = 0; u < 2; ++u) {
        v16bf bb = load_fragB(Km + base + (size_t)((tn0 + u) * 16) * Dd + k0, Dd, lane);
        acc[u] = wmma_bf16(a, bb, acc[u]);
      }
    }
    const float scale = 0.08838834764831845f;  // 1/sqrt(128)
    const int half = lane >> 4, nl = lane & 15;
#pragma unroll
    for (int u = 0; u < 2; ++u) {
      const int n = (tn0 + u) * 16 + nl;
#pragma unroll
      for (int r = 0; r < 8; ++r) {
        const int m = tm * 16 + half * 8 + r;
        sc[m][n] = acc[u][r] * scale + bias64[m * 64 + n];
      }
    }
  }
  __syncthreads();

  // softmax: one thread per row
  if (tid < 64) {
    const int m = tid;
    float mx = -3.4e38f;
    for (int j = 0; j < 64; ++j) mx = fmaxf(mx, sc[m][j]);
    float s = 0.f;
    for (int j = 0; j < 64; ++j) { float e = __expf(sc[m][j] - mx); sc[m][j] = e; s += e; }
    const float inv = 1.f / s;
    for (int j = 0; j < 64; ++j) attS[m][j] = (bf16)(sc[m][j] * inv);
  }
  __syncthreads();

  // ctx = attn @ V : 4x8 tile grid; wave w owns (tm = w&3, tn = 4*(w>>2)+{0..3})
  {
    const int tm = w & 3, tnb = (w >> 2) * 4;
    v8f acc[4] = {};
#pragma unroll
    for (int k0 = 0; k0 < 64; k0 += 32) {
      v16bf a = load_fragA(&attS[tm * 16][0] + k0, 72, lane);
#pragma unroll
      for (int u = 0; u < 4; ++u) {
        v16bf bb = load_fragB(&vt[(tnb + u) * 16][0] + k0, 72, lane);
        acc[u] = wmma_bf16(a, bb, acc[u]);
      }
    }
    const int half = lane >> 4, nl = lane & 15;
#pragma unroll
    for (int u = 0; u < 4; ++u) {
      const int n = (tnb + u) * 16 + nl;
#pragma unroll
      for (int r = 0; r < 8; ++r) {
        const int m = tm * 16 + half * 8 + r;
        Ctx[base + (size_t)m * Dd + n] = (bf16)acc[u][r];
      }
    }
  }
}

// ---------------------------------------------------------------------------
// Fused residual + LayerNorm: y = LN(X[row] + T[row]) * g + b.
// Writes f32 (residual stream) and bf16 (next GEMM A-operand) copies.
// wave32 shfl_xor reduction + one 8-partial LDS pass.
// ---------------------------------------------------------------------------
__global__ __launch_bounds__(256) void ln_kernel(
    const float* __restrict__ X, const float* __restrict__ T,
    const float* __restrict__ g, const float* __restrict__ bta,
    float* __restrict__ Out, bf16* __restrict__ Outb) {
  const int tid = threadIdx.x;
  const size_t row = blockIdx.x;
  const float* xr = X + row * Dd;
  const float* tr = T + row * Dd;
  float v[3], s = 0.f, s2 = 0.f;
#pragma unroll
  for (int i = 0; i < 3; ++i) {
    const int c = tid + i * 256;
    v[i] = xr[c] + tr[c];
    s += v[i]; s2 += v[i] * v[i];
  }
#pragma unroll
  for (int m = 16; m >= 1; m >>= 1) {
    s  += __shfl_xor(s, m, 32);
    s2 += __shfl_xor(s2, m, 32);
  }
  __shared__ float ps[8], ps2[8];
  if ((tid & 31) == 0) { ps[tid >> 5] = s; ps2[tid >> 5] = s2; }
  __syncthreads();
  float ts = 0.f, ts2 = 0.f;
#pragma unroll
  for (int i = 0; i < 8; ++i) { ts += ps[i]; ts2 += ps2[i]; }
  const float mu  = ts * (1.f / Dd);
  const float var = ts2 * (1.f / Dd) - mu * mu;
  const float inv = rsqrtf(var + 1e-5f);
  float* orow  = Out + row * Dd;
  bf16*  obrow = Outb + row * Dd;
#pragma unroll
  for (int i = 0; i < 3; ++i) {
    const int c = tid + i * 256;
    const float y = (v[i] - mu) * inv * g[c] + bta[c];
    orow[c]  = y;
    obrow[c] = (bf16)y;
  }
}

// Weight convert+transpose: Wt[n*K + k] = bf16(W[k*N + n])
__global__ void wt_convert(const float* __restrict__ W, bf16* __restrict__ Wt,
                           int K, int N) {
  const size_t i = (size_t)blockIdx.x * 256 + threadIdx.x;
  if (i >= (size_t)K * N) return;
  const size_t k = i % (size_t)K, n = i / (size_t)K;
  Wt[n * (size_t)K + k] = (bf16)W[k * (size_t)N + n];
}

__global__ void f2b(const float* __restrict__ in, bf16* __restrict__ out, size_t n) {
  const size_t i = (size_t)blockIdx.x * 256 + threadIdx.x;
  if (i < n) out[i] = (bf16)in[i];
}

__global__ void bias_build(const int* __restrict__ sp, const float* __restrict__ tab,
                           float* __restrict__ out) {
  const int i = blockIdx.x * 256 + threadIdx.x;
  if (i < 64 * 64) out[i] = tab[sp[i]];
}

// ---------------------------------------------------------------------------
extern "C" void kernel_launch(void* const* d_in, const int* in_sizes, int n_in,
                              void* d_out, int out_size, void* d_ws, size_t ws_size,
                              hipStream_t stream) {
  (void)in_sizes; (void)n_in; (void)out_size; (void)ws_size;
  const float* x   = (const float*)d_in[0];
  const int*   sp  = (const int*)d_in[1];
  const float* tab = (const float*)d_in[2];
  const float* Wq  = (const float*)d_in[3];
  const float* Wk  = (const float*)d_in[4];
  const float* Wv  = (const float*)d_in[5];
  const float* Wo  = (const float*)d_in[6];
  const float* bq  = (const float*)d_in[7];
  const float* bk  = (const float*)d_in[8];
  const float* bv  = (const float*)d_in[9];
  const float* bo  = (const float*)d_in[10];
  const float* W1  = (const float*)d_in[11];
  const float* b1  = (const float*)d_in[12];
  const float* W2  = (const float*)d_in[13];
  const float* b2  = (const float*)d_in[14];
  const float* l1s = (const float*)d_in[15];
  const float* l1b = (const float*)d_in[16];
  const float* l2s = (const float*)d_in[17];
  const float* l2b = (const float*)d_in[18];

  constexpr size_t BSD = (size_t)Mrows * Dd;
  constexpr size_t BSF = (size_t)Mrows * FFf;
  constexpr size_t WDD = (size_t)Dd * Dd;
  constexpr size_t WDF = (size_t)Dd * FFf;
  constexpr size_t WPL = 4 * WDD + 2 * WDF;  // bf16 weight elements per layer

  char* ws = (char*)d_ws;
  size_t off = 0;
  auto carve = [&](size_t bytes) {
    char* p = ws + off;
    off += (bytes + 255) & ~(size_t)255;
    return p;
  };
  float* X    = (float*)carve(BSD * 4);     // f32 residual stream
  float* T    = (float*)carve(BSD * 4);     // f32 pre-LN projection output
  bf16*  Xb   = (bf16*)carve(BSD * 2);      // bf16 activation (GEMM A operand)
  bf16*  Qb   = (bf16*)carve(BSD * 2);      // reused as ctx output
  bf16*  Kb   = (bf16*)carve(BSD * 2);
  bf16*  Vb   = (bf16*)carve(BSD * 2);
  bf16*  Hb   = (bf16*)carve(BSF * 2);
  bf16*  Wbf  = (bf16*)carve(3 * WPL * 2);
  float* bias64 = (float*)carve(64 * 64 * 4);

  // prepass: bias table, bf16 input activations, bf16 transposed weights
  bias_build<<<16, 256, 0, stream>>>(sp, tab, bias64);
  f2b<<<(unsigned)((BSD + 255) / 256), 256, 0, stream>>>(x, Xb, BSD);
  for (int l = 0; l < 3; ++l) {
    bf16* wl = Wbf + (size_t)l * WPL;
    const unsigned gdd = (unsigned)((WDD + 255) / 256);
    const unsigned gdf = (unsigned)((WDF + 255) / 256);
    wt_convert<<<gdd, 256, 0, stream>>>(Wq + l * WDD, wl,                 Dd, Dd);
    wt_convert<<<gdd, 256, 0, stream>>>(Wk + l * WDD, wl + WDD,           Dd, Dd);
    wt_convert<<<gdd, 256, 0, stream>>>(Wv + l * WDD, wl + 2 * WDD,       Dd, Dd);
    wt_convert<<<gdd, 256, 0, stream>>>(Wo + l * WDD, wl + 3 * WDD,       Dd, Dd);
    wt_convert<<<gdf, 256, 0, stream>>>(W1 + l * WDF, wl + 4 * WDD,       Dd, FFf);
    wt_convert<<<gdf, 256, 0, stream>>>(W2 + l * WDF, wl + 4 * WDD + WDF, FFf, Dd);
  }

  const dim3 blk(256);
  const dim3 gD(Dd / 128, Mrows / 128);    // (6, 512)
  const dim3 gF(FFf / 128, Mrows / 128);   // (24, 512)

  for (int l = 0; l < 3; ++l) {
    bf16* wl = Wbf + (size_t)l * WPL;
    const float* xin = (l == 0) ? x : X;   // f32 residual source
    gemm_kernel<0><<<gD, blk, 0, stream>>>(Xb, wl,           bq + l * Dd, Qb, Mrows, Dd, Dd);
    gemm_kernel<0><<<gD, blk, 0, stream>>>(Xb, wl + WDD,     bk + l * Dd, Kb, Mrows, Dd, Dd);
    gemm_kernel<0><<<gD, blk, 0, stream>>>(Xb, wl + 2 * WDD, bv + l * Dd, Vb, Mrows, Dd, Dd);
    attn_kernel<<<Bb * Hh, blk, 0, stream>>>(Qb, Kb, Vb, bias64, Qb);
    gemm_kernel<2><<<gD, blk, 0, stream>>>(Qb, wl + 3 * WDD, bo + l * Dd, T, Mrows, Dd, Dd);
    ln_kernel<<<Mrows, blk, 0, stream>>>(xin, T, l1s + l * Dd, l1b + l * Dd, X, Xb);
    gemm_kernel<1><<<gF, blk, 0, stream>>>(Xb, wl + 4 * WDD, b1 + l * FFf, Hb, Mrows, FFf, Dd);
    gemm_kernel<2><<<gD, blk, 0, stream>>>(Hb, wl + 4 * WDD + WDF, b2 + l * Dd, T, Mrows, Dd, FFf);
    float* lnout = (l == 2) ? (float*)d_out : X;
    ln_kernel<<<Mrows, blk, 0, stream>>>(X, T, l2s + l * Dd, l2b + l * Dd, lnout, Xb);
  }
}
